// Network_40931038331288
// MI455X (gfx1250) — matrix operand order repaired and checked by reading, not compiled, not run on hardware
//
#include <hip/hip_runtime.h>

typedef _Float16 h2  __attribute__((ext_vector_type(2)));
typedef _Float16 h8  __attribute__((ext_vector_type(8)));
typedef _Float16 h16 __attribute__((ext_vector_type(16)));
typedef float    f4  __attribute__((ext_vector_type(4)));
typedef float    f8  __attribute__((ext_vector_type(8)));

#define T_LEN 500
#define NIN   1024
#define NOUT  512

// ---------------------------------------------------------------------------
// Kernel 1:  Y[b][t][o] = sum_i W[o][i] * X[b][i][t]
// Block tile: 128o x 128t, 256 threads = 8 waves, K-step 32 (f16 WMMA).
// ---------------------------------------------------------------------------
__global__ __launch_bounds__(256)
void snn_gemm(const float* __restrict__ X, const float* __restrict__ W,
              float* __restrict__ Y)
{
    __shared__ __align__(16) _Float16 Wt[128 * 32]; // [o_local][k], row stride 32 halves
    __shared__ __align__(16) _Float16 Xt[128 * 32]; // [t_local][k], row stride 32 halves

    const int tid = threadIdx.x;
    const int b   = blockIdx.z;
    const int ob  = blockIdx.y * 128;
    const int tb  = blockIdx.x * 128;

    const int wave   = tid >> 5;
    const int lane   = tid & 31;
    const int wave_o = wave & 1;   // o offset 64*wave_o
    const int wave_t = wave >> 1;  // t offset 32*wave_t

    // W-tile loader mapping: row wr (0..127), 16-col segment wseg (0 or 16)
    const int wr   = tid >> 1;
    const int wseg = (tid & 1) * 16;
    // X-tile loader mapping: i-pair xp (0..15), 8-t segment base xts
    const int xp   = tid & 15;
    const int xts  = (tid >> 4) * 8;

    f8 acc[4][2] = {};

    for (int i0 = 0; i0 < NIN; i0 += 32) {
        __syncthreads();

        // ---- load W tile -> LDS (f32 -> f16), layout [o][k] ----
        {
            const float* wp = W + (size_t)(ob + wr) * NIN + i0 + wseg;
            f4 a  = *(const f4*)(wp);
            f4 bq = *(const f4*)(wp + 4);
            f4 c  = *(const f4*)(wp + 8);
            f4 dq = *(const f4*)(wp + 12);
            h8 lo, hi;
#pragma unroll
            for (int q = 0; q < 4; ++q) {
                lo[q]     = (_Float16)a[q];
                lo[q + 4] = (_Float16)bq[q];
                hi[q]     = (_Float16)c[q];
                hi[q + 4] = (_Float16)dq[q];
            }
            *(h8*)&Wt[wr * 32 + wseg]     = lo;
            *(h8*)&Wt[wr * 32 + wseg + 8] = hi;
        }

        // ---- load X tile -> LDS transposed (f32 -> f16), layout [t][k] ----
        {
            const int trow = tb + xts;
            const float* x0 = X + ((size_t)b * NIN + (i0 + 2 * xp)) * T_LEN + trow;
            const float* x1 = x0 + T_LEN;
            float v0[8], v1[8];
            if (trow + 8 <= T_LEN) {
                f4 a0 = *(const f4*)x0, b0 = *(const f4*)(x0 + 4);
                f4 a1 = *(const f4*)x1, b1 = *(const f4*)(x1 + 4);
#pragma unroll
                for (int q = 0; q < 4; ++q) {
                    v0[q] = a0[q]; v0[q + 4] = b0[q];
                    v1[q] = a1[q]; v1[q + 4] = b1[q];
                }
            } else {
#pragma unroll
                for (int j = 0; j < 8; ++j) {
                    const int t = trow + j;
                    v0[j] = (t < T_LEN) ? x0[j] : 0.0f;
                    v1[j] = (t < T_LEN) ? x1[j] : 0.0f;
                }
            }
#pragma unroll
            for (int j = 0; j < 8; ++j) {
                h2 pk;
                pk[0] = (_Float16)v0[j];
                pk[1] = (_Float16)v1[j];
                *(h2*)&Xt[(xts + j) * 32 + 2 * xp] = pk;
            }
        }

        __syncthreads();

        // ---- WMMA compute: 4 (o) x 2 (t) tiles of 16x16, K = 32 ----
        const _Float16* wbase = &Wt[(wave_o * 64) * 32];
        const _Float16* xbase = &Xt[(wave_t * 32) * 32];
        const int arow = lane & 15;
        const int koff = (lane < 16) ? 0 : 8;   // A half-K offset
        const int boff = (lane < 16) ? 0 : 8;   // B half-K offset

        h16 afrag[4];
#pragma unroll
        for (int ko = 0; ko < 4; ++ko) {
            const _Float16* p = wbase + (ko * 16 + arow) * 32 + koff;
            h8 lo = *(const h8*)p;
            h8 hi = *(const h8*)(p + 16);
            afrag[ko] = __builtin_shufflevector(lo, hi,
                0, 1, 2, 3, 4, 5, 6, 7, 8, 9, 10, 11, 12, 13, 14, 15);
        }
        h16 bfrag[2];
#pragma unroll
        for (int kt = 0; kt < 2; ++kt) {
            const _Float16* p = xbase + (kt * 16 + arow) * 32 + boff;
            h8 lo = *(const h8*)p;
            h8 hi = *(const h8*)(p + 16);
            bfrag[kt] = __builtin_shufflevector(lo, hi,
                0, 1, 2, 3, 4, 5, 6, 7, 8, 9, 10, 11, 12, 13, 14, 15);
        }
#pragma unroll
        for (int ko = 0; ko < 4; ++ko)
#pragma unroll
            for (int kt = 0; kt < 2; ++kt)
                acc[ko][kt] = __builtin_amdgcn_wmma_f32_16x16x32_f16(
                    false, afrag[ko], false, bfrag[kt],
                    (short)0, acc[ko][kt], false, false);
    }

    // ---- epilogue: D layout -> Y[b][t][o] (o contiguous per lane) ----
    const int n     = lane & 15;          // N (= t) within 16-tile
    const int mhalf = (lane >> 4) * 8;    // M base (0 or 8)
#pragma unroll
    for (int ko = 0; ko < 4; ++ko) {
#pragma unroll
        for (int kt = 0; kt < 2; ++kt) {
            const int tg = tb + wave_t * 32 + kt * 16 + n;
            if (tg < T_LEN) {
                const int og = ob + wave_o * 64 + ko * 16 + mhalf;
                float* dst = Y + ((size_t)b * T_LEN + tg) * NOUT + og;
                f4 lo = __builtin_shufflevector(acc[ko][kt], acc[ko][kt], 0, 1, 2, 3);
                f4 hi = __builtin_shufflevector(acc[ko][kt], acc[ko][kt], 4, 5, 6, 7);
                *(f4*)dst       = lo;
                *(f4*)(dst + 4) = hi;
            }
        }
    }
}

// ---------------------------------------------------------------------------
// Kernel 2: fused PSP-conv + refractory scan via exact 2-state alpha-kernel
// recurrences. One thread per (b, o). Reads Y[b][t][o] coalesced; stages
// spikes in LDS so global writes are wide per-thread contiguous runs.
// ---------------------------------------------------------------------------
__global__ __launch_bounds__(256)
void snn_scan(const float* __restrict__ Y, float* __restrict__ out)
{
    __shared__ float stage[50 * 256];
    const int tid = threadIdx.x;
    const int b   = blockIdx.y;
    const int o   = blockIdx.x * 256 + tid;

    const float* yb = Y + (size_t)b * T_LEN * NOUT + o;
    float* obase    = out + ((size_t)b * NOUT + o) * T_LEN;

    const float d  = 0.77880078307140486825f;   // exp(-1/tau), tau = 4
    const float c1 = 0.67957045711476130884f;   // e/tau          (PSP scale)
    const float c2 = -13.591409142295226177f;   // -2*theta*e/tau (refractory)

    float P = 0.f, Q = 0.f, A = 0.f, Bs = 0.f;

    for (int tc = 0; tc < 10; ++tc) {
#pragma unroll 5
        for (int j = 0; j < 50; ++j) {
            const int t = tc * 50 + j;
            const float y = yb[(size_t)t * NOUT];
            Q  = d * (Q + P);       // PSP state (excludes y[t], srm[0]=0)
            P  = d * P + y;
            Bs = d * (Bs + A);      // refractory state (excludes s[t])
            A  = d * A;
            const float u = c1 * Q + c2 * Bs;
            const float s = (u >= 10.0f) ? 1.0f : 0.0f;
            A += s;
            stage[j * 256 + tid] = s;
        }
        // per-thread contiguous writeback (no cross-thread LDS use -> no barrier)
        float* dst = obase + tc * 50;
#pragma unroll
        for (int j = 0; j < 50; ++j)
            dst[j] = stage[j * 256 + tid];
    }
}

// ---------------------------------------------------------------------------
extern "C" void kernel_launch(void* const* d_in, const int* in_sizes, int n_in,
                              void* d_out, int out_size, void* d_ws, size_t ws_size,
                              hipStream_t stream)
{
    (void)in_sizes; (void)n_in; (void)out_size; (void)ws_size;
    const float* X = (const float*)d_in[0];   // (64,1024,1,1,500) f32 spikes
    const float* W = (const float*)d_in[1];   // (512,1024) f32
    float* out = (float*)d_out;               // (64,512,1,1,500) f32
    float* Y   = (float*)d_ws;                // workspace: 64*500*512 f32 = 65.5 MB

    dim3 g1(4, 4, 64);                        // (t-tiles, o-tiles, batch)
    snn_gemm<<<g1, 256, 0, stream>>>(X, W, Y);
    dim3 g2(2, 64);                           // (o-halves, batch)
    snn_scan<<<g2, 256, 0, stream>>>(Y, out);
}
